// E8_codebook_45990509806216
// MI455X (gfx1250) — compile-verified, matrix operand order checked by reading, not run on hardware
//
#include <hip/hip_runtime.h>

typedef __attribute__((ext_vector_type(16))) _Float16     v16h;
typedef __attribute__((ext_vector_type(8)))  _Float16     v8h;
typedef __attribute__((ext_vector_type(8)))  float        v8f;
typedef __attribute__((ext_vector_type(8)))  unsigned int v8u;
typedef __attribute__((ext_vector_type(4)))  unsigned int v4u;

#define WAVES_PER_BLOCK 8
#define THREADS (WAVES_PER_BLOCK * 32)

// One wave32 processes a 32-row tile of X (two 16-row A matrices; lanes 16..31's
// rows are shuffled into A1 so both WMMAs have full-EXEC wave-uniform operands).
// The f16 codebook (B-layout) + per-column bias -0.5*|g|^2 are staged once per
// block in LDS; the inner loop is branch-free: ds_load_b128 + ds_load_b32 +
// 2x v_wmma_f32_16x16x32_f16 + argmax update.
__global__ __launch_bounds__(THREADS)
void e8_quant_wmma(const float* __restrict__ X,
                   const float* __restrict__ grid_part,
                   const float* __restrict__ grid_norm,
                   const short* __restrict__ allcombo_idx,  // [128, P] int16
                   const int*   __restrict__ idx_map,       // [256]
                   float* __restrict__ out_vals,            // [N, 8]
                   float* __restrict__ out_idx,             // [N] (int16 values as float)
                   int N, int numTiles, int P)
{
    extern __shared__ unsigned char smem[];
    const int PT   = (P + 15) >> 4;
    const int cols = PT << 4;
    // LDS layout: [0,16) zero pad | [16, 16+cols*16) B cols (8 f16 each) | cols*4 bias floats
    float* cvals = (float*)(smem + 16 + (size_t)cols * 16);

    // ---- Stage codebook into LDS in WMMA-B layout (once per block) ----
    if (threadIdx.x == 0) *(v4u*)smem = (v4u){0u, 0u, 0u, 0u};
    for (int i = threadIdx.x; i < cols; i += THREADS) {
        _Float16* bc = (_Float16*)(smem + 16 + (size_t)i * 16);
        if (i < P) {
            const float4* gr = (const float4*)(grid_part + (size_t)i * 8);
            float4 lo = gr[0], hi = gr[1];
            bc[0] = (_Float16)lo.x; bc[1] = (_Float16)lo.y;
            bc[2] = (_Float16)lo.z; bc[3] = (_Float16)lo.w;
            bc[4] = (_Float16)hi.x; bc[5] = (_Float16)hi.y;
            bc[6] = (_Float16)hi.z; bc[7] = (_Float16)hi.w;
            cvals[i] = -0.5f * grid_norm[i];
        } else {
#pragma unroll
            for (int j = 0; j < 8; ++j) bc[j] = (_Float16)0.0f;
            cvals[i] = -3.0e38f;                 // padding columns can never win
        }
    }
    __syncthreads();

    const int lane = threadIdx.x & 31;
    const int wv   = threadIdx.x >> 5;
    const int tile = blockIdx.x * WAVES_PER_BLOCK + wv;
    if (tile >= numTiles) return;                // wave-uniform (after the barrier)

    const int half  = lane >> 4;                 // 0: K=0..15 slots, 1: K=16..31 slots
    const int l16   = lane & 15;
    const int myrow = tile * 32 + lane;          // every lane owns one output row
    const bool rowOk = (myrow < N);

    // ---- Load this lane's row; canonicalize sign/parity ----
    float mask[8];
#pragma unroll
    for (int i = 0; i < 8; ++i) mask[i] = 1.0f;
    int packed = 0;
    v8h hOwn = {};
    if (rowOk) {
        const float4* xr = (const float4*)(X + (size_t)myrow * 8);
        float4 xlo = xr[0], xhi = xr[1];
        float xv[8] = {xlo.x, xlo.y, xlo.z, xlo.w, xhi.x, xhi.y, xhi.z, xhi.w};
        int negbits = 0;
#pragma unroll
        for (int i = 0; i < 8; ++i) negbits |= (xv[i] < 0.0f) ? (1 << i) : 0;
        const int   odd   = __popc(negbits) & 1;
        const float sign0 = odd ? -1.0f : 1.0f;
#pragma unroll
        for (int i = 0; i < 8; ++i) mask[i] = ((negbits >> i) & 1) ? -1.0f : 1.0f;
        mask[0] *= sign0;
        packed = (negbits & 0xFE) | ((negbits ^ odd) & 1);
        float xp[8];
#pragma unroll
        for (int i = 0; i < 8; ++i) xp[i] = fabsf(xv[i]);
        xp[0] *= sign0;
#pragma unroll
        for (int i = 0; i < 8; ++i) hOwn[i] = (_Float16)xp[i];
    }
    const v4u hu = __builtin_bit_cast(v4u, hOwn);

    // ---- A0 (rows 0..15): lanes 0..15 use own data. A1 (rows 16..31): shuffle. ----
    v8u a0w = { half ? 0u : hu[0], half ? 0u : hu[1],
                half ? 0u : hu[2], half ? 0u : hu[3], 0u, 0u, 0u, 0u };
    unsigned s0 = (unsigned)__shfl((int)hu[0], l16 + 16, 32);
    unsigned s1 = (unsigned)__shfl((int)hu[1], l16 + 16, 32);
    unsigned s2 = (unsigned)__shfl((int)hu[2], l16 + 16, 32);
    unsigned s3 = (unsigned)__shfl((int)hu[3], l16 + 16, 32);
    v8u a1w = { half ? 0u : s0, half ? 0u : s1,
                half ? 0u : s2, half ? 0u : s3, 0u, 0u, 0u, 0u };
    const v16h a0 = __builtin_bit_cast(v16h, a0w);
    const v16h a1 = __builtin_bit_cast(v16h, a1w);

    // ---- Running argmax state ----
    float best0[8], best1[8];
    int   bcol0[8], bcol1[8];
#pragma unroll
    for (int r = 0; r < 8; ++r) {
        best0[r] = -3.0e38f; bcol0[r] = 0;
        best1[r] = -3.0e38f; bcol1[r] = 0;
    }

    // Branch-free per-lane LDS cursors: lanes 16..31 keep reading the zero pad.
    unsigned bOff = half ? 0u : (16u + (unsigned)l16 * 16u);
    const unsigned bStride = half ? 0u : 256u;
    unsigned cOff = 16u + (unsigned)cols * 16u + (unsigned)l16 * 4u;

    for (int ct = 0; ct < PT; ++ct) {
        const v4u  bu   = *(const v4u*)(smem + bOff);
        const float cval = *(const float*)(smem + cOff);
        bOff += bStride;
        cOff += 64u;

        v8u bw = { bu[0], bu[1], bu[2], bu[3], 0u, 0u, 0u, 0u };
        const v16h b = __builtin_bit_cast(v16h, bw);

        v8f c;
#pragma unroll
        for (int r = 0; r < 8; ++r) c[r] = cval;

        // D = A*B + C  ->  x.g - 0.5*|g|^2  (argmax-equivalent to 2 x.g - |g|^2)
        v8f d0 = __builtin_amdgcn_wmma_f32_16x16x32_f16(
            false, a0, false, b, (short)0, c, false, false);
        v8f d1 = __builtin_amdgcn_wmma_f32_16x16x32_f16(
            false, a1, false, b, (short)0, c, false, false);

        const int col = ct * 16 + l16;
#pragma unroll
        for (int r = 0; r < 8; ++r) {
            float t0 = d0[r];
            if (t0 > best0[r]) { best0[r] = t0; bcol0[r] = col; }
            float t1 = d1[r];
            if (t1 > best1[r]) { best1[r] = t1; bcol1[r] = col; }
        }
    }

    // ---- Argmax reduction across the 16 column-lanes of each half-wave ----
#pragma unroll
    for (int m = 1; m <= 8; m <<= 1) {
#pragma unroll
        for (int r = 0; r < 8; ++r) {
            float os0 = __shfl_xor(best0[r], m, 32);
            int   oc0 = __shfl_xor(bcol0[r], m, 32);
            if (os0 > best0[r] || (os0 == best0[r] && oc0 < bcol0[r])) {
                best0[r] = os0; bcol0[r] = oc0;
            }
            float os1 = __shfl_xor(best1[r], m, 32);
            int   oc1 = __shfl_xor(bcol1[r], m, 32);
            if (os1 > best1[r] || (os1 == best1[r] && oc1 < bcol1[r])) {
                best1[r] = os1; bcol1[r] = oc1;
            }
        }
    }

    // ---- Route winners: rows 0..7 in lanes 0..15 of set0, 8..15 in lanes 16..31
    //      of set0, 16..23 / 24..31 likewise in set1. Lane L owns row L. ----
    int myCol = 0;
#pragma unroll
    for (int r = 0; r < 8; ++r) {
        int c0a = __shfl(bcol0[r], 0, 32);
        int c0b = __shfl(bcol0[r], 16, 32);
        int c1a = __shfl(bcol1[r], 0, 32);
        int c1b = __shfl(bcol1[r], 16, 32);
        if (lane == r)      myCol = c0a;
        if (lane == r + 8)  myCol = c0b;
        if (lane == r + 16) myCol = c1a;
        if (lane == r + 24) myCol = c1b;
    }

    // ---- Emit outputs: vals = grid_part[best] * mask ; idx via double gather ----
    if (rowOk) {
        const float4* gr = (const float4*)(grid_part + (size_t)myCol * 8);
        float4 glo = gr[0], ghi = gr[1];
        float gv[8] = {glo.x, glo.y, glo.z, glo.w, ghi.x, ghi.y, ghi.z, ghi.w};
        float* ov = out_vals + (size_t)myrow * 8;
#pragma unroll
        for (int i = 0; i < 8; ++i) ov[i] = gv[i] * mask[i];
        const int prow = idx_map[packed & 255];
        out_idx[myrow] = (float)allcombo_idx[(size_t)prow * (size_t)P + (size_t)myCol];
    }
}

extern "C" void kernel_launch(void* const* d_in, const int* in_sizes, int n_in,
                              void* d_out, int out_size, void* d_ws, size_t ws_size,
                              hipStream_t stream) {
    const float* X          = (const float*)d_in[0];  // [N, 8] f32
    const float* grid_part  = (const float*)d_in[1];  // [P, 8] f32
    const float* grid_norm  = (const float*)d_in[2];  // [P]    f32
    // d_in[3] = int_map (powers of two; folded into the bit-pack, unused)
    const short* allcombo   = (const short*)d_in[4];  // [128, P] int16
    const int*   idx_map    = (const int*)d_in[5];    // [256] int32

    const int N = in_sizes[0] / 8;
    const int P = in_sizes[1] / 8;

    float* out_vals = (float*)d_out;                  // first N*8 floats
    float* out_idx  = (float*)d_out + (size_t)N * 8;  // next N floats

    const int numTiles = (N + 31) / 32;               // 32 rows per wave
    const int blocks   = (numTiles + WAVES_PER_BLOCK - 1) / WAVES_PER_BLOCK;

    const int PT   = (P + 15) / 16;
    const int cols = PT * 16;
    const size_t shmem = 16 + (size_t)cols * 16 + (size_t)cols * 4;

    hipLaunchKernelGGL(e8_quant_wmma, dim3(blocks), dim3(THREADS), shmem, stream,
                       X, grid_part, grid_norm, allcombo, idx_map,
                       out_vals, out_idx, N, numTiles, P);
}
